// MLPBlock_13116830122494
// MI455X (gfx1250) — matrix operand order repaired and checked by reading, not compiled, not run on hardware
//
#include <hip/hip_runtime.h>
#include <stdint.h>
#include <stddef.h>

// ---------------- problem constants (fixed by reference) ----------------
#define T_TOK 2048
#define HDIM  512
#define IDIM  512
#define NEXP  8
#define KTOP  4

#define EPS_RMS 1e-6f
#define LIMIT   7.0f
#define W_LOAD  0.01f
#define W_IMP   0.01f
#define K_ALPHA 1.702f

// ---------------- CDNA5 WMMA / TDM types ----------------
typedef __attribute__((ext_vector_type(16))) __bf16 v16bf;
typedef __attribute__((ext_vector_type(8)))  float  v8f;
typedef __attribute__((ext_vector_type(4)))  float  f32x4;
typedef __attribute__((ext_vector_type(4)))  unsigned int u32x4;
typedef __attribute__((ext_vector_type(8)))  int i32x8;
typedef __attribute__((ext_vector_type(4)))  int i32x4;

union BfFrag { v16bf v; f32x4 q[2]; };

#define KS       128                 // K-slab width (elements)
#define LDS_LDA  136                 // 128 elems + TDM pad of 4 DWORDs (16B) per 256B row
#define A_SLAB_B (64 * LDS_LDA * 2)  // 17408 B
#define B_SLAB_B (128 * LDS_LDA * 2) // 34816 B

__device__ __forceinline__ unsigned short f32_bf16(float f) {
  unsigned int u = __float_as_uint(f);
  unsigned int r = u + 0x7fffu + ((u >> 16) & 1u);   // round-to-nearest-even
  return (unsigned short)(r >> 16);
}

// ---------------- TDM: 2D tile (tile_rows x 128 bf16) global -> LDS ----------------
// Pads 4 DWORDs after every 64 DWORDs streamed => LDS row stride = 136 elements.
__device__ __forceinline__ void tdm_load_tile(unsigned short* lds,
                                              const unsigned short* gtile,
                                              unsigned tile_rows, unsigned ld) {
#if __has_builtin(__builtin_amdgcn_tensor_load_to_lds)
  unsigned long long ga = (unsigned long long)(uintptr_t)gtile;
  u32x4 g0;
  g0[0] = 1u;                                        // count=1, user mode
  g0[1] = (unsigned)(uintptr_t)lds;                  // lds_addr (low 32 bits = LDS offset)
  g0[2] = (unsigned)(ga & 0xffffffffu);              // global_addr[31:0]
  g0[3] = (unsigned)((ga >> 32) & 0x01ffffffu)       // global_addr[56:32]
        | (2u << 30);                                // type = 2 ("image")
  i32x8 g1;
  // wg_mask=0 | data_size=1(2B)<<16 | pad_enable<<20 | pad_interval=5(64DW)<<22 | pad_amount=3(4DW)<<25
  g1[0] = (int)((1u << 16) | (1u << 20) | (5u << 22) | (3u << 25));
  g1[1] = (int)((ld & 0xffffu) << 16);                                   // tensor_dim0 lo16 @ [63:48]
  g1[2] = (int)(((ld >> 16) & 0xffffu) | ((tile_rows & 0xffffu) << 16)); // dim0 hi16 | tensor_dim1 lo16
  g1[3] = (int)(((tile_rows >> 16) & 0xffffu) | ((unsigned)KS << 16));   // dim1 hi16 | tile_dim0=128
  g1[4] = (int)(tile_rows & 0xffffu);                                    // tile_dim1 | tile_dim2=0
  g1[5] = (int)ld;                                                       // tensor_dim0_stride lo32
  g1[6] = 0;                                                             // stride hi | dim1_stride
  g1[7] = 0;
  i32x4 z4 = {0, 0, 0, 0};
  i32x8 z8 = {0, 0, 0, 0, 0, 0, 0, 0};
  // 6-arg form (clang-23 / therock-10.0): (g0, g1, g2, g3, g4, cpol)
  __builtin_amdgcn_tensor_load_to_lds(g0, g1, z4, z4, z8, 0);
#else
  // fallback: synchronous single-wave copy with identical padded LDS layout
  const int lane = threadIdx.x & 31;
  for (unsigned idx = (unsigned)lane; idx < tile_rows * (unsigned)KS; idx += 32) {
    unsigned r = idx >> 7, c = idx & (KS - 1);
    lds[r * LDS_LDA + c] = gtile[(size_t)r * ld + c];
  }
#endif
}

template <int N>
__device__ __forceinline__ void tdm_wait() {
#if __has_builtin(__builtin_amdgcn_s_wait_tensorcnt)
  __builtin_amdgcn_s_wait_tensorcnt(N);   // immediate required
#endif
}

// ---------------- kernel: fp32 -> bf16 conversion ----------------
__global__ void convert_bf16_k(const float* __restrict__ src,
                               unsigned short* __restrict__ dst, int n) {
  int i = blockIdx.x * blockDim.x + threadIdx.x;
  int stride = gridDim.x * blockDim.x;
  for (; i < n; i += stride) dst[i] = f32_bf16(src[i]);
}

__global__ void init_stats_k(float* __restrict__ s) {
  if (threadIdx.x < 48) s[threadIdx.x] = 0.0f;
}

// ---------------- kernel: RMSNorm + gate + top-4 + aux stats ----------------
__global__ __launch_bounds__(256) void rms_gate_k(
    const float* __restrict__ x, const float* __restrict__ scale,
    const float* __restrict__ gw, const float* __restrict__ gb,
    unsigned short* __restrict__ t_bf, float* __restrict__ coeff,
    float* __restrict__ psum, float* __restrict__ isum,
    float* __restrict__ cnts) {
  __shared__ float t_s[HDIM];
  __shared__ float red[8];
  __shared__ float logit_s[NEXP];

  const int t   = blockIdx.x;
  const int tid = threadIdx.x;
  const float* xr = x + (size_t)t * HDIM;

  float x0 = xr[tid], x1 = xr[tid + 256];
  float ss = x0 * x0 + x1 * x1;
  for (int m = 16; m > 0; m >>= 1) ss += __shfl_xor(ss, m, 32);
  if ((tid & 31) == 0) red[tid >> 5] = ss;
  __syncthreads();
  if (tid == 0) {
    float tot = 0.f;
    for (int i = 0; i < 8; i++) tot += red[i];
    red[0] = rsqrtf(tot / (float)HDIM + EPS_RMS);
  }
  __syncthreads();
  float inv = red[0];
  float t0 = x0 * inv * scale[tid];
  float t1 = x1 * inv * scale[tid + 256];
  t_s[tid] = t0; t_s[tid + 256] = t1;
  t_bf[(size_t)t * HDIM + tid]       = f32_bf16(t0);
  t_bf[(size_t)t * HDIM + tid + 256] = f32_bf16(t1);
  __syncthreads();

  const int e = tid >> 5, lane = tid & 31;
  const float* wr = gw + (size_t)e * HDIM;
  float acc = 0.f;
  for (int i = lane; i < HDIM; i += 32) acc += t_s[i] * wr[i];
  for (int m = 16; m > 0; m >>= 1) acc += __shfl_xor(acc, m, 32);
  if (lane == 0) logit_s[e] = acc + gb[e];
  __syncthreads();

  if (tid == 0) {
    float lg[NEXP], p[NEXP];
    float mx = -1e30f;
    for (int i = 0; i < NEXP; i++) { lg[i] = logit_s[i]; mx = fmaxf(mx, lg[i]); }
    float sum = 0.f;
    for (int i = 0; i < NEXP; i++) { p[i] = __expf(lg[i] - mx); sum += p[i]; }
    for (int i = 0; i < NEXP; i++) p[i] /= sum;
    bool used[NEXP]; for (int i = 0; i < NEXP; i++) used[i] = false;
    int idx[KTOP]; float w[KTOP]; float wsum = 0.f;
    for (int k = 0; k < KTOP; k++) {
      int best = 0; float bv = -1.f;
      for (int i = 0; i < NEXP; i++)
        if (!used[i] && p[i] > bv) { bv = p[i]; best = i; }
      used[best] = true; idx[k] = best; w[k] = bv; wsum += bv;
    }
    float c[NEXP]; for (int i = 0; i < NEXP; i++) c[i] = 0.f;
    for (int k = 0; k < KTOP; k++) c[idx[k]] += w[k] / wsum;
    for (int i = 0; i < NEXP; i++) coeff[(size_t)t * NEXP + i] = c[i];
    for (int i = 0; i < NEXP; i++) { atomicAdd(&psum[i], p[i]); atomicAdd(&isum[i], lg[i]); }
    for (int k = 0; k < KTOP; k++) atomicAdd(&cnts[k * NEXP + idx[k]], 1.0f);
  }
}

// ---------------- LDS fragment loaders (CDNA5 16x16x32 bf16 layouts) ----------------
__device__ __forceinline__ void lds_load_a(BfFrag& f, const unsigned short* row, int kk, int hh) {
  f.q[0] = *(const f32x4*)(row + kk + hh * 8);
  f.q[1] = *(const f32x4*)(row + kk + 16 + hh * 8);
}
__device__ __forceinline__ void lds_load_b(BfFrag& f, const unsigned short* row, int kk, int hh) {
  f.q[0] = *(const f32x4*)(row + kk + hh * 8);
  f.q[1] = *(const f32x4*)(row + kk + 16 + hh * 8);
}

#define WMMA_BF16(A, B, C) \
  __builtin_amdgcn_wmma_f32_16x16x32_bf16(false, (A), false, (B), (short)0, (C), false, false)

// ---------------- kernel: GEMM1 (h = t @ W1^T + b1) fused bias/clip/SwiGLU ----------------
// block tile 64(M) x 128(N); 8 waves (2x4) of 32x32; TDM double-buffered K-slabs
#define BM1 64
#define BN1 128
__global__ __launch_bounds__(256) void gemm1_swiglu_k(
    const unsigned short* __restrict__ tbf, const unsigned short* __restrict__ w1,
    const float* __restrict__ b1, unsigned short* __restrict__ abf) {
  __shared__ __align__(16) char smem[2 * A_SLAB_B + 2 * B_SLAB_B];  // 104448 B
  unsigned short* As[2] = {(unsigned short*)smem,
                           (unsigned short*)(smem + A_SLAB_B)};
  unsigned short* Bs[2] = {(unsigned short*)(smem + 2 * A_SLAB_B),
                           (unsigned short*)(smem + 2 * A_SLAB_B + B_SLAB_B)};
  float* hs = (float*)smem;  // 64 x 132 f32, reused AFTER the K loop (aliases slabs)

  const int e    = blockIdx.z;
  const int row0 = blockIdx.y * BM1;
  const int col0 = blockIdx.x * BN1;
  const int wid  = threadIdx.x >> 5, lane = threadIdx.x & 31;
  const int wm = wid & 1, wn = wid >> 1;
  const int mlane = lane & 15, hh = lane >> 4;

  const unsigned short* W  = w1 + (size_t)e * (2 * IDIM) * HDIM;
  const unsigned short* tA = tbf + (size_t)row0 * HDIM;   // 64 rows
  const unsigned short* tB = W + (size_t)col0 * HDIM;     // 128 rows

  const int NS = HDIM / KS;  // 4 slabs
  if (threadIdx.x < 32) {
    tdm_load_tile(As[0], tA, BM1, HDIM);
    tdm_load_tile(Bs[0], tB, BN1, HDIM);
  }

  v8f c00 = {}, c01 = {}, c10 = {}, c11 = {};
  for (int s = 0; s < NS; s++) {
    if (threadIdx.x < 32) {
      if (s + 1 < NS) {
        tdm_load_tile(As[(s + 1) & 1], tA + (s + 1) * KS, BM1, HDIM);
        tdm_load_tile(Bs[(s + 1) & 1], tB + (s + 1) * KS, BN1, HDIM);
        tdm_wait<2>();   // slab s complete; slab s+1 still in flight
      } else {
        tdm_wait<0>();   // final slab complete
      }
    }
    __syncthreads();
    const int b = s & 1;
    const unsigned short* ar0 = As[b] + (wm * 32 + mlane) * LDS_LDA;
    const unsigned short* ar1 = As[b] + (wm * 32 + 16 + mlane) * LDS_LDA;
    const unsigned short* br0 = Bs[b] + (wn * 32 + mlane) * LDS_LDA;
    const unsigned short* br1 = Bs[b] + (wn * 32 + 16 + mlane) * LDS_LDA;
#pragma unroll
    for (int kk = 0; kk < KS; kk += 32) {
      BfFrag a0, a1, b0, b1f;
      lds_load_a(a0, ar0, kk, hh);
      lds_load_a(a1, ar1, kk, hh);
      lds_load_b(b0, br0, kk, hh);
      lds_load_b(b1f, br1, kk, hh);
      c00 = WMMA_BF16(a0.v, b0.v,  c00);
      c01 = WMMA_BF16(a0.v, b1f.v, c01);
      c10 = WMMA_BF16(a1.v, b0.v,  c10);
      c11 = WMMA_BF16(a1.v, b1f.v, c11);
    }
    __syncthreads();   // readers done before next slab overwrites this buffer
  }

  // bias + clip, stage h in LDS (aliases slab buffers; protected by barrier above)
  const float* bias = b1 + (size_t)e * (2 * IDIM) + col0 + wn * 32;
  const float bc0 = bias[mlane];
  const float bc1 = bias[16 + mlane];
#pragma unroll
  for (int v = 0; v < 8; v++) {
    int r0 = wm * 32 + v + 8 * hh;
    int r1 = r0 + 16;
    hs[r0 * 132 + wn * 32 + mlane]      = fminf(fmaxf(c00[v] + bc0, -LIMIT), LIMIT);
    hs[r0 * 132 + wn * 32 + 16 + mlane] = fminf(fmaxf(c01[v] + bc1, -LIMIT), LIMIT);
    hs[r1 * 132 + wn * 32 + mlane]      = fminf(fmaxf(c10[v] + bc0, -LIMIT), LIMIT);
    hs[r1 * 132 + wn * 32 + 16 + mlane] = fminf(fmaxf(c11[v] + bc1, -LIMIT), LIMIT);
  }
  __syncthreads();

  // SwiGLU over interleaved column pairs; write bf16 activations [E][T][I]
  const int acol0 = col0 >> 1;
  for (int idx = threadIdx.x; idx < BM1 * (BN1 / 2); idx += 256) {
    int r = idx >> 6;
    int c = idx & 63;
    float g = hs[r * 132 + 2 * c];
    float l = hs[r * 132 + 2 * c + 1];
    float av = g * (1.0f / (1.0f + __expf(-K_ALPHA * g))) + (l + 1.0f);
    abf[(size_t)e * T_TOK * IDIM + (size_t)(row0 + r) * IDIM + acol0 + c] = f32_bf16(av);
  }
}

// ---------------- kernel: GEMM2 over all experts + coeff-combine + residual ----------------
#define BM2 64
#define BN2 128
__global__ __launch_bounds__(256) void gemm2_combine_k(
    const unsigned short* __restrict__ abf, const unsigned short* __restrict__ w2,
    const float* __restrict__ b2, const float* __restrict__ coeff,
    const float* __restrict__ x, float* __restrict__ out) {
  __shared__ __align__(16) char smem[2 * A_SLAB_B + 2 * B_SLAB_B];
  __shared__ float cf[BM2][NEXP];
  unsigned short* As[2] = {(unsigned short*)smem,
                           (unsigned short*)(smem + A_SLAB_B)};
  unsigned short* Bs[2] = {(unsigned short*)(smem + 2 * A_SLAB_B),
                           (unsigned short*)(smem + 2 * A_SLAB_B + B_SLAB_B)};

  const int row0 = blockIdx.y * BM2;
  const int col0 = blockIdx.x * BN2;
  const int tid  = threadIdx.x;
  for (int i = tid; i < BM2 * NEXP; i += 256)
    cf[i >> 3][i & 7] = coeff[(size_t)(row0 + (i >> 3)) * NEXP + (i & 7)];

  const int wid = tid >> 5, lane = tid & 31;
  const int wm = wid & 1, wn = wid >> 1;
  const int mlane = lane & 15, hh = lane >> 4;
  const int NS = IDIM / KS;

  v8f f00 = {}, f01 = {}, f10 = {}, f11 = {};
  for (int e = 0; e < NEXP; e++) {
    const unsigned short* tA = abf + (size_t)e * T_TOK * IDIM + (size_t)row0 * IDIM;
    const unsigned short* tB = w2 + (size_t)e * HDIM * IDIM + (size_t)col0 * IDIM;
    if (threadIdx.x < 32) {
      tdm_load_tile(As[0], tA, BM2, IDIM);
      tdm_load_tile(Bs[0], tB, BN2, IDIM);
    }
    v8f c00 = {}, c01 = {}, c10 = {}, c11 = {};
    for (int s = 0; s < NS; s++) {
      if (threadIdx.x < 32) {
        if (s + 1 < NS) {
          tdm_load_tile(As[(s + 1) & 1], tA + (s + 1) * KS, BM2, IDIM);
          tdm_load_tile(Bs[(s + 1) & 1], tB + (s + 1) * KS, BN2, IDIM);
          tdm_wait<2>();
        } else {
          tdm_wait<0>();
        }
      }
      __syncthreads();
      const int b = s & 1;
      const unsigned short* ar0 = As[b] + (wm * 32 + mlane) * LDS_LDA;
      const unsigned short* ar1 = As[b] + (wm * 32 + 16 + mlane) * LDS_LDA;
      const unsigned short* br0 = Bs[b] + (wn * 32 + mlane) * LDS_LDA;
      const unsigned short* br1 = Bs[b] + (wn * 32 + 16 + mlane) * LDS_LDA;
#pragma unroll
      for (int kk = 0; kk < KS; kk += 32) {
        BfFrag a0, a1, b0, b1f;
        lds_load_a(a0, ar0, kk, hh);
        lds_load_a(a1, ar1, kk, hh);
        lds_load_b(b0, br0, kk, hh);
        lds_load_b(b1f, br1, kk, hh);
        c00 = WMMA_BF16(a0.v, b0.v,  c00);
        c01 = WMMA_BF16(a0.v, b1f.v, c01);
        c10 = WMMA_BF16(a1.v, b0.v,  c10);
        c11 = WMMA_BF16(a1.v, b1f.v, c11);
      }
      __syncthreads();
    }
    const float bc0 = b2[(size_t)e * HDIM + col0 + wn * 32 + mlane];
    const float bc1 = b2[(size_t)e * HDIM + col0 + wn * 32 + 16 + mlane];
#pragma unroll
    for (int v = 0; v < 8; v++) {
      float cw0 = cf[wm * 32 + v + 8 * hh][e];
      float cw1 = cf[wm * 32 + 16 + v + 8 * hh][e];
      f00[v] += cw0 * (c00[v] + bc0);
      f01[v] += cw0 * (c01[v] + bc1);
      f10[v] += cw1 * (c10[v] + bc0);
      f11[v] += cw1 * (c11[v] + bc1);
    }
  }

#pragma unroll
  for (int v = 0; v < 8; v++) {
    int r0 = row0 + wm * 32 + v + 8 * hh, r1 = r0 + 16;
    int cA = col0 + wn * 32 + mlane, cB = cA + 16;
    out[(size_t)r0 * HDIM + cA] = f00[v] + x[(size_t)r0 * HDIM + cA];
    out[(size_t)r0 * HDIM + cB] = f01[v] + x[(size_t)r0 * HDIM + cB];
    out[(size_t)r1 * HDIM + cA] = f10[v] + x[(size_t)r1 * HDIM + cA];
    out[(size_t)r1 * HDIM + cB] = f11[v] + x[(size_t)r1 * HDIM + cB];
  }
}

// ---------------- kernel: aux loss ----------------
__global__ void aux_loss_k(const float* __restrict__ psum, const float* __restrict__ isum,
                           const float* __restrict__ cnts, float* __restrict__ out_aux) {
  if (threadIdx.x == 0 && blockIdx.x == 0) {
    float loadsum = 0.f;
    for (int e = 0; e < NEXP; e++) {
      float P = psum[e] / (float)T_TOK;
      float d = 0.f;
      for (int k = 0; k < KTOP; k++)
        d += (float)(1 << (KTOP - k)) * cnts[k * NEXP + e];
      d /= (float)(T_TOK * KTOP);
      loadsum += P * d;
    }
    float loss_load = W_LOAD * (float)NEXP * loadsum;
    float mean = 0.f;
    for (int e = 0; e < NEXP; e++) mean += isum[e];
    mean /= (float)NEXP;
    float var = 0.f;
    for (int e = 0; e < NEXP; e++) { float d = isum[e] - mean; var += d * d; }
    var /= (float)(NEXP - 1);
    float cv = sqrtf(var) / (mean + 1e-6f);
    *out_aux = loss_load + W_IMP * cv * cv;
  }
}

// ---------------- workspace layout (bytes) ----------------
#define OFF_T     ((size_t)0)
#define OFF_W1    ((size_t)2097152)
#define OFF_W2    ((size_t)10485760)
#define OFF_A     ((size_t)14680064)
#define OFF_COEFF ((size_t)31457280)
#define OFF_STATS ((size_t)31522816)

extern "C" void kernel_launch(void* const* d_in, const int* in_sizes, int n_in,
                              void* d_out, int out_size, void* d_ws, size_t ws_size,
                              hipStream_t stream) {
  const float* x          = (const float*)d_in[0];
  const float* norm_scale = (const float*)d_in[1];
  const float* gate_w     = (const float*)d_in[2];
  const float* gate_b     = (const float*)d_in[3];
  const float* mlp1_w     = (const float*)d_in[4];
  const float* mlp1_b     = (const float*)d_in[5];
  const float* mlp2_w     = (const float*)d_in[6];
  const float* mlp2_b     = (const float*)d_in[7];

  float* out = (float*)d_out;
  float* aux = out + (size_t)T_TOK * HDIM;

  char* ws = (char*)d_ws;
  unsigned short* t_bf  = (unsigned short*)(ws + OFF_T);
  unsigned short* w1_bf = (unsigned short*)(ws + OFF_W1);
  unsigned short* w2_bf = (unsigned short*)(ws + OFF_W2);
  unsigned short* a_bf  = (unsigned short*)(ws + OFF_A);
  float* coeff = (float*)(ws + OFF_COEFF);
  float* stats = (float*)(ws + OFF_STATS);
  float* psum = stats;
  float* isum = stats + 8;
  float* cnts = stats + 16;

  convert_bf16_k<<<1024, 256, 0, stream>>>(mlp1_w, w1_bf, NEXP * 2 * IDIM * HDIM);
  convert_bf16_k<<<1024, 256, 0, stream>>>(mlp2_w, w2_bf, NEXP * HDIM * IDIM);
  init_stats_k<<<1, 64, 0, stream>>>(stats);

  rms_gate_k<<<T_TOK, 256, 0, stream>>>(x, norm_scale, gate_w, gate_b,
                                        t_bf, coeff, psum, isum, cnts);

  gemm1_swiglu_k<<<dim3((2 * IDIM) / BN1, T_TOK / BM1, NEXP), 256, 0, stream>>>(
      t_bf, w1_bf, mlp1_b, a_bf);

  gemm2_combine_k<<<dim3(HDIM / BN2, T_TOK / BM2), 256, 0, stream>>>(
      a_bf, w2_bf, mlp2_b, coeff, x, out);

  aux_loss_k<<<1, 32, 0, stream>>>(psum, isum, cnts, aux);
}
